// SlidingWindowRotaryMultiheadAttention_73959336837191
// MI455X (gfx1250) — compile-verified, matrix operand order checked by reading, not run on hardware
//
#include <hip/hip_runtime.h>
#include <hip/hip_bf16.h>
#include <math.h>

typedef __bf16 bf16_t;
typedef __attribute__((ext_vector_type(16))) __bf16 v16bf;
typedef __attribute__((ext_vector_type(8)))  __bf16 v8bf;
typedef __attribute__((ext_vector_type(8)))  float  v8f;
typedef int v4i_t __attribute__((vector_size(16)));   // matches builtin's V4i

#define NH     16
#define DH     32
#define E_DIM  512
#define B_SZ   2
#define T_SZ   8
#define HW_    256
#define L2_    512
#define S_SZ   4096
#define WIN    4

// ---- CDNA5 async global->LDS path (guarded; falls back to sync b128 copies)
#ifndef ASYNC_LDS
#  if defined(__has_builtin)
#    if __has_builtin(__builtin_amdgcn_global_load_async_to_lds_b128)
#      define ASYNC_LDS 1
#    endif
#  endif
#endif
#ifndef ASYNC_LDS
#  define ASYNC_LDS 0
#endif

#if ASYNC_LDS
#  if defined(__has_builtin) && __has_builtin(__builtin_amdgcn_s_wait_asynccnt)
#    define WAIT_ASYNC(n) __builtin_amdgcn_s_wait_asynccnt(n)
#  else
#    define WAIT_ASYNC(n) asm volatile("s_wait_asynccnt %0" ::"n"(n) : "memory")
#  endif
#else
#  define WAIT_ASYNC(n) ((void)0)
#endif

#define GLB_AS __attribute__((address_space(1)))
#define LDS_AS __attribute__((address_space(3)))

// copy 16 bytes (8 bf16) global -> LDS
static __device__ __forceinline__ void cp16(const bf16_t* g, bf16_t* l) {
#if ASYNC_LDS
  __builtin_amdgcn_global_load_async_to_lds_b128(
      (GLB_AS v4i_t*)g, (LDS_AS v4i_t*)l, 0, 0);
#else
  *(int4*)l = *(const int4*)g;
#endif
}

// f32 -> bf16 round-to-nearest-even (bit trick)
static __device__ __forceinline__ bf16_t f2bf(float x) {
  union { float f; unsigned u; } v; v.f = x;
  unsigned r = v.u + 0x7FFFu + ((v.u >> 16) & 1u);
  unsigned short h = (unsigned short)(r >> 16);
  return __builtin_bit_cast(bf16_t, h);
}
static __device__ __forceinline__ float bf2f(bf16_t h) {
  unsigned v = ((unsigned)__builtin_bit_cast(unsigned short, h)) << 16;
  return __builtin_bit_cast(float, v);
}

// ---------------------------------------------------------------------------
// bulk f32 -> bf16 conversion (n must be a multiple of 4)
// ---------------------------------------------------------------------------
__global__ __launch_bounds__(256)
void cvt_kernel(const float* __restrict__ src, bf16_t* __restrict__ dst, int n4)
{
  int i = blockIdx.x * 256 + threadIdx.x;
  if (i < n4) {
    float4 v = ((const float4*)src)[i];
    bf16_t o[4] = { f2bf(v.x), f2bf(v.y), f2bf(v.z), f2bf(v.w) };
    uint2 u; __builtin_memcpy(&u, o, 8);
    ((uint2*)dst)[i] = u;
  }
}

// ---------------------------------------------------------------------------
// C = A(MxK,bf16) * W^T(NxK,bf16) + bias(f32); out f32 or bf16.
// 128x128x32 tile, 8 waves (2x4), wave = 4x2 WMMA tiles.
// Double-buffered async global->LDS staging (4 async b128 per thread/tile).
// ---------------------------------------------------------------------------
#define GM 128
#define GN 128
#define GK 32
#define LDS_S 40   // padded LDS row stride (elements) -> conflict-free b128 reads

__global__ __launch_bounds__(256)
void gemm_bf16_wmma(const bf16_t* __restrict__ A, int lda,
                    const bf16_t* __restrict__ Wt,   // N x K row-major
                    const float* __restrict__ bias,
                    void* __restrict__ Cout, int N, int K, int out_f32)
{
  __shared__ bf16_t sA[2][GM * LDS_S];
  __shared__ bf16_t sB[2][GN * LDS_S];

  const int tid  = threadIdx.x;
  const int wave = tid >> 5, lane = tid & 31;
  const int half = lane >> 4, l16 = lane & 15;
  const int wm = wave >> 2, wn = wave & 3;
  const int mBase = blockIdx.y * GM;
  const int nBase = blockIdx.x * GN;

  // staging assignment: thread copies 32B of one row of A and one row of B
  const int srow = tid >> 1;            // 0..127
  const int soff = (tid & 1) * 16;      // element offset 0 or 16
  const bf16_t* gA = A  + (size_t)(mBase + srow) * lda + soff;
  const bf16_t* gB = Wt + (size_t)(nBase + srow) * K   + soff;
  bf16_t* lA0 = &sA[0][srow * LDS_S + soff];
  bf16_t* lA1 = &sA[1][srow * LDS_S + soff];
  bf16_t* lB0 = &sB[0][srow * LDS_S + soff];
  bf16_t* lB1 = &sB[1][srow * LDS_S + soff];

  v8f zero = {};
  v8f acc[4][2];
#pragma unroll
  for (int i = 0; i < 4; ++i)
#pragma unroll
    for (int j = 0; j < 2; ++j) acc[i][j] = zero;

  const int nk = K / GK;
  // prologue: stage tile 0 into buffer 0
  cp16(gA, lA0); cp16(gA + 8, lA0 + 8);
  cp16(gB, lB0); cp16(gB + 8, lB0 + 8);

  for (int i = 0; i < nk; ++i) {
    const int cur = i & 1;
    if (i + 1 < nk) {      // issue next tile into the other buffer
      const bf16_t* ga = gA + (size_t)(i + 1) * GK;
      const bf16_t* gb = gB + (size_t)(i + 1) * GK;
      bf16_t* la = cur ? lA0 : lA1;
      bf16_t* lb = cur ? lB0 : lB1;
      cp16(ga, la); cp16(ga + 8, la + 8);
      cp16(gb, lb); cp16(gb + 8, lb + 8);
      WAIT_ASYNC(4);       // tile i complete (its 4 asyncs retired in order)
    } else {
      WAIT_ASYNC(0);
    }
    __syncthreads();

    v16bf bfr[2];
#pragma unroll
    for (int nt = 0; nt < 2; ++nt) {
      const bf16_t* bp = &sB[cur][(wn * 32 + nt * 16 + l16) * LDS_S];
      ((v8bf*)&bfr[nt])[0] = *(const v8bf*)(bp + half * 8);
      ((v8bf*)&bfr[nt])[1] = *(const v8bf*)(bp + 16 + half * 8);
    }
#pragma unroll
    for (int mt = 0; mt < 4; ++mt) {
      v16bf af;
      const bf16_t* ap = &sA[cur][(wm * 64 + mt * 16 + l16) * LDS_S];
      ((v8bf*)&af)[0] = *(const v8bf*)(ap + half * 8);
      ((v8bf*)&af)[1] = *(const v8bf*)(ap + 16 + half * 8);
#pragma unroll
      for (int nt = 0; nt < 2; ++nt)
        acc[mt][nt] = __builtin_amdgcn_wmma_f32_16x16x32_bf16(
            false, af, false, bfr[nt], (short)0, acc[mt][nt], false, false);
    }
    __syncthreads();
  }

  // epilogue (C layout: lanes 0-15 -> N=l16 M=0..7; lanes 16-31 -> M=8..15)
#pragma unroll
  for (int mt = 0; mt < 4; ++mt)
#pragma unroll
    for (int nt = 0; nt < 2; ++nt) {
      int col = nBase + wn * 32 + nt * 16 + l16;
      float bv = bias[col];
#pragma unroll
      for (int r = 0; r < 8; ++r) {
        int row = mBase + wm * 64 + mt * 16 + half * 8 + r;
        float val = acc[mt][nt][r] + bv;
        if (out_f32) ((float*)Cout)[(size_t)row * N + col] = val;
        else         ((bf16_t*)Cout)[(size_t)row * N + col] = f2bf(val);
      }
    }
}

// ---------------------------------------------------------------------------
// RoPE + token reorder, bf16 in/out.
// ---------------------------------------------------------------------------
__global__ __launch_bounds__(256)
void rope_kernel(const bf16_t* __restrict__ qkv,
                 bf16_t* __restrict__ qout, bf16_t* __restrict__ kout)
{
  const int blk = blockIdx.x;          // b*S + s
  const int b = blk >> 12;
  const int s = blk & 4095;
  const int j  = threadIdx.x;          // pair 0..255
  const int jj = j & 15;
  const int head = j >> 4;
  const float LN = 9.210340371976184f; // ln(10000)

  float ang;
  int dstRow;
  if (s < 2048) {                      // 1D rope
    int t = s >> 8, l = s & 255;
    float f = __expf(-LN * (float)jj * (1.0f / 16.0f));
    ang = (float)l * f;
    dstRow = (b * T_SZ + t) * L2_ + l;
  } else {                             // 2D rope
    int s2 = s - 2048;
    int t = s2 >> 8, l2 = s2 & 255;
    int hh = l2 >> 4, ww = l2 & 15;
    float f;
    if (jj < 8) { f = __expf(-LN * (float)(2 * jj) * (1.0f / 16.0f)); ang = (float)hh * f; }
    else { int m = jj - 8; f = __expf(-LN * (float)(2 * m + 1) * (1.0f / 16.0f)); ang = (float)ww * f; }
    dstRow = (b * T_SZ + t) * L2_ + HW_ + l2;
  }
  float c, sn;
  __sincosf(ang, &sn, &c);

  const size_t srow = (size_t)blk * 1536;
  const int col = head * DH + 2 * jj;
  unsigned qp = *(const unsigned*)&qkv[srow + col];
  unsigned kp = *(const unsigned*)&qkv[srow + 512 + col];
  float qre = bf2f(__builtin_bit_cast(bf16_t, (unsigned short)(qp & 0xFFFF)));
  float qim = bf2f(__builtin_bit_cast(bf16_t, (unsigned short)(qp >> 16)));
  float kre = bf2f(__builtin_bit_cast(bf16_t, (unsigned short)(kp & 0xFFFF)));
  float kim = bf2f(__builtin_bit_cast(bf16_t, (unsigned short)(kp >> 16)));

  const size_t drow = (size_t)dstRow * E_DIM;
  bf16_t q2[2] = { f2bf(qre * c - qim * sn), f2bf(qre * sn + qim * c) };
  bf16_t k2[2] = { f2bf(kre * c - kim * sn), f2bf(kre * sn + kim * c) };
  unsigned uq, uk;
  __builtin_memcpy(&uq, q2, 4); __builtin_memcpy(&uk, k2, 4);
  *(unsigned*)&qout[drow + col] = uq;
  *(unsigned*)&kout[drow + col] = uk;
}

// te_proj[i] = temporal_embed[i] @ Wk^T (no bias), f32
__global__ __launch_bounds__(256)
void teproj_kernel(const float* __restrict__ te, const float* __restrict__ ipw,
                   float* __restrict__ tp)
{
  const int i = blockIdx.x;
  for (int e = threadIdx.x; e < E_DIM; e += 256) {
    float a = 0.f;
    const float* w = &ipw[(size_t)(E_DIM + e) * E_DIM];
    const float* t = &te[i * E_DIM];
    for (int k = 0; k < E_DIM; ++k) a += t[k] * w[k];
    tp[i * E_DIM + e] = a;
  }
}

// ---------------------------------------------------------------------------
// Flash-style attention, bf16 in/out. Grid (qtile=4, head=16, bt=16), 8 waves.
// te folded into per-row score bias: q.(k+te_w)^T = q.k^T + (q.te_w) scalar.
// Padded window rows are just bk / bv bias vectors (bf16).
// ---------------------------------------------------------------------------
__global__ __launch_bounds__(256)
void attn_kernel(const bf16_t* __restrict__ qh, const bf16_t* __restrict__ kproj,
                 const bf16_t* __restrict__ vproj, const float* __restrict__ tp,
                 const bf16_t* __restrict__ ipbb, bf16_t* __restrict__ o)
{
  __shared__ bf16_t sQ[128 * LDS_S];
  __shared__ bf16_t sK[32 * LDS_S];
  __shared__ bf16_t sV[32 * LDS_S];        // TRANSPOSED: [dcol][key]
  __shared__ bf16_t sP[8][16 * LDS_S];
  __shared__ float  sQTE[8][16][4];        // per wave-row: q . te_proj[win]

  const int tid = threadIdx.x, wave = tid >> 5, lane = tid & 31;
  const int half = lane >> 4, l16 = lane & 15;
  const int q0   = blockIdx.x * 128;
  const int head = blockIdx.y;
  const int bt   = blockIdx.z;
  const int b = bt >> 3, t = bt & 7;
  const int hc = head * DH;

  { // stage Q (128x32 bf16): 2 x 16B per thread, async
    int r = tid >> 1, coff = (tid & 1) * 16;
    const bf16_t* g = &qh[(size_t)(bt * L2_ + q0 + r) * E_DIM + hc + coff];
    bf16_t* l = &sQ[r * LDS_S + coff];
    cp16(g, l); cp16(g + 8, l + 8);
  }
  WAIT_ASYNC(0);
  __syncthreads();

  v16bf qf;
  {
    const bf16_t* qp = &sQ[(wave * 16 + l16) * LDS_S];
    ((v8bf*)&qf)[0] = *(const v8bf*)(qp + half * 8);
    ((v8bf*)&qf)[1] = *(const v8bf*)(qp + 16 + half * 8);
  }
  { // q . te_proj bias: each lane does 2 of its wave's 16rows x 4wins dots
    const int w0 = half * 2;
    const bf16_t* qrow = &sQ[(wave * 16 + l16) * LDS_S];
    float a0 = 0.f, a1 = 0.f;
    for (int kx = 0; kx < DH; ++kx) {
      float qv = bf2f(qrow[kx]);
      a0 += qv * tp[(w0 + 0) * E_DIM + hc + kx];
      a1 += qv * tp[(w0 + 1) * E_DIM + hc + kx];
    }
    sQTE[wave][l16][w0]     = a0;
    sQTE[wave][l16][w0 + 1] = a1;
  }

  float rmax[8], rsum[8], qb[8];
#pragma unroll
  for (int r = 0; r < 8; ++r) { rmax[r] = -1e30f; rsum[r] = 0.f; qb[r] = 0.f; }
  v8f zero = {};
  v8f o0 = zero, o1 = zero;
  const float scale = 0.17677669529663687f; // 1/sqrt(32)

  for (int ci = 0; ci < 64; ++ci) {
    const int win = ci >> 4;
    const int lp0 = (ci * 32) & 511;
    const int tsrc = t - (WIN - 1) + win;
    const bool valid = (tsrc >= 0);

    if (tid < 128) { // stage K chunk (32x32 bf16) pure copy, async
      int kr = tid >> 2, coff = (tid & 3) * 8;
      const bf16_t* g = valid
          ? &kproj[(size_t)((b * T_SZ + tsrc) * L2_ + lp0 + kr) * E_DIM + hc + coff]
          : &ipbb[E_DIM + hc + coff];                 // bk
      cp16(g, &sK[kr * LDS_S + coff]);
    }
    { // stage V transposed (manual 2B scatter)
      int vr = tid >> 3, dc0 = (tid & 7) * 4;
      const bf16_t* g = valid
          ? &vproj[(size_t)(b * S_SZ + tsrc * L2_ + lp0 + vr) * E_DIM + hc + dc0]
          : &ipbb[2 * E_DIM + hc + dc0];              // bv
      uint2 u = *(const uint2*)g;
      bf16_t e[4]; __builtin_memcpy(e, &u, 8);
      sV[(dc0 + 0) * LDS_S + vr] = e[0];
      sV[(dc0 + 1) * LDS_S + vr] = e[1];
      sV[(dc0 + 2) * LDS_S + vr] = e[2];
      sV[(dc0 + 3) * LDS_S + vr] = e[3];
    }
    WAIT_ASYNC(0);
    __syncthreads();

    if ((ci & 15) == 0) { // refresh per-row te bias when window changes
#pragma unroll
      for (int r = 0; r < 8; ++r) qb[r] = sQTE[wave][half * 8 + r][win];
    }

    v16bf kf0, kf1, vf0, vf1;
    {
      const bf16_t* kp0 = &sK[l16 * LDS_S];
      ((v8bf*)&kf0)[0] = *(const v8bf*)(kp0 + half * 8);
      ((v8bf*)&kf0)[1] = *(const v8bf*)(kp0 + 16 + half * 8);
      const bf16_t* kp1 = &sK[(16 + l16) * LDS_S];
      ((v8bf*)&kf1)[0] = *(const v8bf*)(kp1 + half * 8);
      ((v8bf*)&kf1)[1] = *(const v8bf*)(kp1 + 16 + half * 8);
      const bf16_t* vp0 = &sV[l16 * LDS_S];
      ((v8bf*)&vf0)[0] = *(const v8bf*)(vp0 + half * 8);
      ((v8bf*)&vf0)[1] = *(const v8bf*)(vp0 + 16 + half * 8);
      const bf16_t* vp1 = &sV[(16 + l16) * LDS_S];
      ((v8bf*)&vf1)[0] = *(const v8bf*)(vp1 + half * 8);
      ((v8bf*)&vf1)[1] = *(const v8bf*)(vp1 + 16 + half * 8);
    }
    v8f s0 = zero, s1 = zero;
    s0 = __builtin_amdgcn_wmma_f32_16x16x32_bf16(false, qf, false, kf0, (short)0, s0, false, false);
    s1 = __builtin_amdgcn_wmma_f32_16x16x32_bf16(false, qf, false, kf1, (short)0, s1, false, false);

    bf16_t* pw = &sP[wave][0];
#pragma unroll
    for (int r = 0; r < 8; ++r) {
      float a  = (s0[r] + qb[r]) * scale;
      float bb = (s1[r] + qb[r]) * scale;
      float mx = fmaxf(a, bb);
#pragma unroll
      for (int m = 1; m < 16; m <<= 1) mx = fmaxf(mx, __shfl_xor(mx, m, 32));
      float mnew  = fmaxf(rmax[r], mx);
      float alpha = __expf(rmax[r] - mnew);
      float p0 = __expf(a - mnew);
      float p1 = __expf(bb - mnew);
      float ps = p0 + p1;
#pragma unroll
      for (int m = 1; m < 16; m <<= 1) ps += __shfl_xor(ps, m, 32);
      rsum[r] = rsum[r] * alpha + ps;
      rmax[r] = mnew;
      o0[r] = o0[r] * alpha;
      o1[r] = o1[r] * alpha;
      int mrow = half * 8 + r;
      pw[mrow * LDS_S + l16]      = f2bf(p0);
      pw[mrow * LDS_S + 16 + l16] = f2bf(p1);
    }
    __syncthreads();

    v16bf pf;
    const bf16_t* pp = &pw[l16 * LDS_S];
    ((v8bf*)&pf)[0] = *(const v8bf*)(pp + half * 8);
    ((v8bf*)&pf)[1] = *(const v8bf*)(pp + 16 + half * 8);
    o0 = __builtin_amdgcn_wmma_f32_16x16x32_bf16(false, pf, false, vf0, (short)0, o0, false, false);
    o1 = __builtin_amdgcn_wmma_f32_16x16x32_bf16(false, pf, false, vf1, (short)0, o1, false, false);
  }

#pragma unroll
  for (int r = 0; r < 8; ++r) {
    float inv = 1.0f / rsum[r];
    int row = bt * L2_ + q0 + wave * 16 + half * 8 + r;
    o[(size_t)row * E_DIM + hc + l16]      = f2bf(o0[r] * inv);
    o[(size_t)row * E_DIM + hc + 16 + l16] = f2bf(o1[r] * inv);
  }
}

// ---------------------------------------------------------------------------
extern "C" void kernel_launch(void* const* d_in, const int* in_sizes, int n_in,
                              void* d_out, int out_size, void* d_ws, size_t ws_size,
                              hipStream_t stream)
{
  const float* hidden = (const float*)d_in[0];   // (2,4096,512)
  const float* ipw    = (const float*)d_in[1];   // (1536,512)
  const float* ipb    = (const float*)d_in[2];   // (1536,)
  const float* opw    = (const float*)d_in[3];   // (512,512)
  const float* opb    = (const float*)d_in[4];   // (512,)
  const float* te     = (const float*)d_in[5];   // (4,512)

  const size_t M = 8192;
  bf16_t* wb = (bf16_t*)d_ws;
  bf16_t* hidb  = wb;                    // 8192*512
  bf16_t* ipwb  = hidb  + M * 512;       // 1536*512
  bf16_t* opwb  = ipwb  + 1536 * 512;    // 512*512
  bf16_t* ipbb  = opwb  + 512 * 512;     // 1536 (+pad)
  bf16_t* qkv   = ipbb  + 2048;          // 8192*1536
  bf16_t* qrope = qkv   + M * 1536;      // 8192*512 (bt,l) order
  bf16_t* krope = qrope + M * 512;
  bf16_t* qhb   = krope + M * 512;
  bf16_t* kprj  = qhb   + M * 512;
  bf16_t* vprj  = kprj  + M * 512;
  bf16_t* obuf  = vprj  + M * 512;
  float*  tpb   = (float*)(obuf + M * 512); // 4*512 f32 (8B-aligned)

  // 0) convert inputs to bf16
  cvt_kernel<<<dim3((8192 * 512 / 4 + 255) / 256), 256, 0, stream>>>(hidden, hidb, 8192 * 512 / 4);
  cvt_kernel<<<dim3((1536 * 512 / 4 + 255) / 256), 256, 0, stream>>>(ipw, ipwb, 1536 * 512 / 4);
  cvt_kernel<<<dim3((512 * 512 / 4 + 255) / 256), 256, 0, stream>>>(opw, opwb, 512 * 512 / 4);
  cvt_kernel<<<dim3(2), 256, 0, stream>>>(ipb, ipbb, 1536 / 4);

  // 1) qkv projection
  gemm_bf16_wmma<<<dim3(1536 / GN, 8192 / GM), 256, 0, stream>>>(
      hidb, 512, ipwb, ipb, qkv, 1536, 512, 0);
  // 2) RoPE + reorder
  rope_kernel<<<dim3(8192), 256, 0, stream>>>(qkv, qrope, krope);
  // 3) temporal-embed projection (tiny, f32)
  teproj_kernel<<<dim3(WIN), 256, 0, stream>>>(te, ipw, tpb);
  // 4) second projections (window duplication folded out by linearity)
  gemm_bf16_wmma<<<dim3(512 / GN, 8192 / GM), 256, 0, stream>>>(
      qrope, 512, ipwb, ipb, qhb, 512, 512, 0);
  gemm_bf16_wmma<<<dim3(512 / GN, 8192 / GM), 256, 0, stream>>>(
      krope, 512, ipwb + 512 * 512, ipb + 512, kprj, 512, 512, 0);
  gemm_bf16_wmma<<<dim3(512 / GN, 8192 / GM), 256, 0, stream>>>(
      qkv + 1024, 1536, ipwb + 1024 * 512, ipb + 1024, vprj, 512, 512, 0);
  // 5) windowed attention
  attn_kernel<<<dim3(4, NH, B_SZ * T_SZ), 256, 0, stream>>>(
      qhb, kprj, vprj, tpb, ipbb, obuf);
  // 6) output projection -> f32 d_out (row order == (B,S,E) flat)
  gemm_bf16_wmma<<<dim3(512 / GN, 8192 / GM), 256, 0, stream>>>(
      obuf, 512, opwb, opb, d_out, 512, 512, 1);
}
